// RowEncoder_28570122453068
// MI455X (gfx1250) — matrix operand order repaired and checked by reading, not compiled, not run on hardware
//
#include <hip/hip_runtime.h>
#include <stdint.h>

// Problem constants (match reference)
#define B_  64
#define HP_ 16
#define WP_ 32
#define C_  512
#define H_  512
#define L_  2
#define T_  32
#define N_  1024          // B*HP sequences
#define G4_ 2048          // 4*H gate width

typedef __attribute__((ext_vector_type(16))) __bf16 v16bf;
typedef __attribute__((ext_vector_type(8)))  __bf16 v8bf;
typedef __attribute__((ext_vector_type(8)))  float  v8f;

// ---------------- workspace layout ----------------
static constexpr size_t WPK_ELEMS  = (size_t)2*2*32*128*512;   // bf16 swizzled weights (16 MB)
static constexpr size_t XBF_ELEMS  = (size_t)N_*T_*C_;         // bf16 features (32 MB)
static constexpr size_t HBUF_ELEMS = (size_t)2*2*2*N_*H_;      // bf16 h double buffer (8 MB)
static constexpr size_t CBUF_ELEMS = (size_t)2*2*N_*H_;        // f32 cell state (8 MB)

static constexpr size_t OFF_WPK  = 0;
static constexpr size_t OFF_XBF  = OFF_WPK  + WPK_ELEMS*2;
static constexpr size_t OFF_HBUF = OFF_XBF  + XBF_ELEMS*2;
static constexpr size_t OFF_CBUF = OFF_HBUF + HBUF_ELEMS*2;
static constexpr size_t OFF_BIAS = OFF_CBUF + CBUF_ELEMS*4;

// -------- CDNA5 async global->LDS copy (16B) + asynccnt waits --------
// Signature discovered via compiler diagnostic:
//   (v4i addrspace(1)* gsrc, v4i addrspace(3)* ldst, imm offset, imm cpol)
typedef int v4i_ __attribute__((vector_size(16)));
typedef __attribute__((address_space(1))) v4i_* gv4i_p;
typedef __attribute__((address_space(3))) v4i_* lv4i_p;

#if defined(__has_builtin) && __has_builtin(__builtin_amdgcn_global_load_async_to_lds_b128)
#define ASYNC_CP16(gsrc, ldst)                                               \
    __builtin_amdgcn_global_load_async_to_lds_b128(                          \
        (gv4i_p)(gsrc), (lv4i_p)(ldst), 0, 0)
#else
#define ASYNC_CP16(gsrc, ldst)                                               \
    asm volatile("global_load_async_to_lds_b128 %0, %1, off"                 \
                 :: "v"((unsigned)(uintptr_t)(ldst)), "v"(gsrc) : "memory")
#endif

__device__ __forceinline__ void wait_async_le4() {
#if defined(__has_builtin) && __has_builtin(__builtin_amdgcn_s_wait_asynccnt)
    __builtin_amdgcn_s_wait_asynccnt(4);
#else
    asm volatile("s_wait_asynccnt 0x4" ::: "memory");
#endif
}
__device__ __forceinline__ void wait_async_le0() {
#if defined(__has_builtin) && __has_builtin(__builtin_amdgcn_s_wait_asynccnt)
    __builtin_amdgcn_s_wait_asynccnt(0);
#else
    asm volatile("s_wait_asynccnt 0x0" ::: "memory");
#endif
}

// ---------------- pre-pass kernels ----------------
__global__ void pack_x_kernel(const float* __restrict__ f, __bf16* __restrict__ x, int n) {
    int i = blockIdx.x * blockDim.x + threadIdx.x;
    if (i < n) x[i] = (__bf16)f[i];
}

// Pack [Wi;Wh] (K=1024, cols=2048) per (dir,layer) into per-lane WMMA B-fragment order.
__global__ void pack_w_kernel(const float* __restrict__ Wi_fw, const float* __restrict__ Wh_fw,
                              const float* __restrict__ Wi_bw, const float* __restrict__ Wh_bw,
                              __bf16* __restrict__ wpk) {
    int i = blockIdx.x * blockDim.x + threadIdx.x;     // 2*2*1024*2048 = 2^23
    if (i >= 2*2*1024*2048) return;
    int col = i & 2047;
    int K   = (i >> 11) & 1023;
    int l   = (i >> 21) & 1;
    int d   = (i >> 22) & 1;
    const float* Wi = d ? Wi_bw : Wi_fw;
    const float* Wh = d ? Wh_bw : Wh_fw;
    float w = (K < 512) ? Wi[((size_t)l*512 + K)*2048 + col]
                        : Wh[((size_t)l*512 + (K - 512))*2048 + col];
    int kt = K >> 5, kk = K & 31;
    int ct = col >> 4, cn = col & 15;
    int lane = cn + (((kk >> 3) & 1) << 4);
    int e = (kk & 7) + ((kk >> 4) << 3);
    size_t off = ((((size_t)(d*2 + l)*32 + kt)*128 + ct)*512) + (size_t)lane*16 + e;
    wpk[off] = (__bf16)w;
}

__global__ void init_state_kernel(const float* __restrict__ pe_fw, const float* __restrict__ pe_bw,
                                  __bf16* __restrict__ hbuf, float* __restrict__ cbuf) {
    int i = blockIdx.x * blockDim.x + threadIdx.x;     // 2*2*N*H = 2^21
    if (i >= 2*2*N_*H_) return;
    int j = i & 511;
    int n = (i >> 9) & 1023;
    int l = (i >> 19) & 1;
    int d = (i >> 20) & 1;
    const float* pe = d ? pe_bw : pe_fw;
    int row = (n & 15) + 1;
    size_t base = (size_t)row*(L_*2*H_) + (size_t)l*2*H_;
    hbuf[(((size_t)(0*2 + d)*2 + l)*N_ + n)*H_ + j] = (__bf16)pe[base + j];
    cbuf[(((size_t)d*2 + l)*N_ + n)*H_ + j] = pe[base + H_ + j];
}

__global__ void make_bias_kernel(const float* __restrict__ bi_fw, const float* __restrict__ bh_fw,
                                 const float* __restrict__ bi_bw, const float* __restrict__ bh_bw,
                                 float* __restrict__ bias) {
    int i = blockIdx.x * blockDim.x + threadIdx.x;     // 2*2*2048
    if (i >= 2*2*G4_) return;
    int col = i & 2047;
    int l = (i >> 11) & 1;
    int d = (i >> 12) & 1;
    const float* bi = d ? bi_bw : bi_fw;
    const float* bh = d ? bh_bw : bh_fw;
    bias[i] = bi[l*G4_ + col] + bh[l*G4_ + col];
}

// ---------------- recurrent layer-step kernel ----------------
// Grid: (8 col-blocks, 16 row-blocks, 2 dirs). Block: 256 threads = 8 waves (4 row x 2 col).
// B tiles staged through double-buffered LDS via async global->LDS copies; all 8 waves
// consume one copy (4x less L2 traffic). A fragments register-double-buffered.
__global__ __launch_bounds__(256)
void lstm_step_kernel(const __bf16* __restrict__ wpk,
                      const __bf16* __restrict__ xbf,
                      __bf16* __restrict__ hbuf,
                      float* __restrict__ cbuf,
                      const float* __restrict__ bias,
                      float* __restrict__ out,
                      int layer, int s)
{
    __shared__ __bf16 smem[2 * 8192];     // 2 x 16KB B staging

    const int cb   = blockIdx.x;          // h-col block (64 cols)
    const int rb   = blockIdx.y;          // row block (64 rows)
    const int dir  = blockIdx.z;
    const int tid  = threadIdx.x;
    const int lane = tid & 31;
    const int wave = tid >> 5;
    const int wr   = wave >> 1;           // 0..3 : 16-row slice
    const int wc   = wave & 1;            // 0..1 : 32-col slice

    const int t    = dir ? (T_ - 1 - s) : s;
    const int rbuf = s & 1;
    const int wbuf = rbuf ^ 1;
    const int dl   = dir*2 + layer;

    // A operand sources: K[0,512) = x_t (layer0) or h^0_t (layer1); K[512,1024) = h_{t-1}
    const __bf16* a1;
    long a1stride;
    if (layer == 0) { a1 = xbf + (size_t)t*C_;                            a1stride = (long)T_*C_; }
    else            { a1 = hbuf + (((size_t)(wbuf*2 + dir)*2 + 0)*N_)*H_; a1stride = H_; }
    const __bf16* a2 = hbuf + (((size_t)(rbuf*2 + dir)*2 + layer)*N_)*H_;

    const __bf16* wbase = wpk + (size_t)dl*32*128*512;

    const int mRow = rb*64 + wr*16 + (lane & 15);
    const int klo  = (lane >> 4) * 8;
    const __bf16* pA0 = a1 + (long)mRow*a1stride + klo;
    const __bf16* pA1 = a2 + (long)mRow*H_ + klo;

    // async B prefetch of k-chunk kt into LDS buffer b (whole WG moves 16KB)
    const int pg = tid >> 6;              // gate group handled by this thread
    const int po = (tid & 63) * 32;       // element offset inside 4KB gate chunk
    const __bf16* pre_g = wbase + (size_t)(pg*32 + cb*4)*512 + po;
    __bf16* pre_l = &smem[(size_t)pg*2048 + po];

    #define PREFETCH_B(kt, b) do {                                        \
        const __bf16* gs = pre_g + (size_t)(kt)*65536;                    \
        __bf16* ls = pre_l + (size_t)(b)*8192;                            \
        ASYNC_CP16(gs +  0, ls +  0);                                     \
        ASYNC_CP16(gs +  8, ls +  8);                                     \
        ASYNC_CP16(gs + 16, ls + 16);                                     \
        ASYNC_CP16(gs + 24, ls + 24);                                     \
    } while (0)

    v8f zero = {};
    v8f acc[4][2];
    #pragma unroll
    for (int g = 0; g < 4; ++g) { acc[g][0] = zero; acc[g][1] = zero; }

    // prologue: prefetch chunk 0, load A fragment 0
    PREFETCH_B(0, 0);
    v8bf lo0 = *(const v8bf*)(pA0);
    v8bf hi0 = *(const v8bf*)(pA0 + 16);
    v16bf aCur = __builtin_shufflevector(lo0, hi0,
        0,1,2,3,4,5,6,7,8,9,10,11,12,13,14,15);

    const __bf16* bfragBase = &smem[0] + (size_t)wc*1024 + (size_t)lane*16;

    #pragma unroll
    for (int kt = 0; kt < 32; ++kt) {
        v16bf aNxt = aCur;
        if (kt < 31) {
            PREFETCH_B(kt + 1, (kt + 1) & 1);
            const int kn = kt + 1;
            const __bf16* p = (kn < 16) ? (pA0 + kn*32) : (pA1 + (kn - 16)*32);
            v8bf alo = *(const v8bf*)p;
            v8bf ahi = *(const v8bf*)(p + 16);
            aNxt = __builtin_shufflevector(alo, ahi,
                0,1,2,3,4,5,6,7,8,9,10,11,12,13,14,15);
            wait_async_le4();             // our 4 copies for chunk kt are done
        } else {
            wait_async_le0();
        }
        __syncthreads();                  // all waves' copies for chunk kt are done

        const __bf16* bb = bfragBase + (size_t)(kt & 1)*8192;
        v16bf bfr[8];
        #pragma unroll
        for (int u = 0; u < 8; ++u) {
            const int g = u >> 1, jt = u & 1;
            bfr[u] = *(const v16bf*)(bb + (size_t)g*2048 + (size_t)jt*512);
        }
        #pragma unroll
        for (int u = 0; u < 8; ++u) {
            const int g = u >> 1, jt = u & 1;
            acc[g][jt] = __builtin_amdgcn_wmma_f32_16x16x32_bf16(
                false, aCur, false, bfr[u], (short)0, acc[g][jt], false, false);
        }
        __syncthreads();                  // buffer (kt&1) free for reuse at kt+2
        aCur = aNxt;
    }
    #undef PREFETCH_B

    // Epilogue: C/D layout -> thread holds (M = r + 8*(lane>=16), N = lane%16)
    const int nInTile = lane & 15;
    const int rowEpi  = rb*64 + wr*16 + ((lane >> 4) << 3);
    const int colB    = cb*64 + wc*32;
    __bf16* hout = hbuf + (((size_t)(wbuf*2 + dir)*2 + layer)*N_)*H_;
    float*  cbp  = cbuf + ((size_t)dl*N_)*H_;
    const float* bp = bias + (size_t)dl*G4_;

    #pragma unroll
    for (int jt = 0; jt < 2; ++jt) {
        const int j = colB + jt*16 + nInTile;
        const float bI = bp[0*H_ + j];
        const float bF = bp[1*H_ + j];
        const float bO = bp[2*H_ + j];
        const float bG = bp[3*H_ + j];
        #pragma unroll
        for (int r = 0; r < 8; ++r) {
            const int n = rowEpi + r;
            float iv = acc[0][jt][r] + bI;
            float fv = acc[1][jt][r] + bF;
            float ov = acc[2][jt][r] + bO;
            float gv = acc[3][jt][r] + bG;
            float si = 1.f / (1.f + __expf(-iv));
            float sf = 1.f / (1.f + __expf(-fv));
            float so = 1.f / (1.f + __expf(-ov));
            float tg = tanhf(gv);
            size_t cidx = (size_t)n*H_ + j;
            float c = sf * cbp[cidx] + si * tg;
            cbp[cidx] = c;
            float h = so * tanhf(c);
            hout[(size_t)n*H_ + j] = (__bf16)h;
            if (layer == 1) {
                size_t o = (((size_t)(n >> 4)*512) + (size_t)(n & 15)*T_ + t)*1024
                           + (size_t)dir*H_ + j;
                out[o] = h;
            }
        }
    }
}

// ---------------- host launcher ----------------
extern "C" void kernel_launch(void* const* d_in, const int* in_sizes, int n_in,
                              void* d_out, int out_size, void* d_ws, size_t ws_size,
                              hipStream_t stream) {
    (void)in_sizes; (void)n_in; (void)out_size; (void)ws_size;
    const float* features = (const float*)d_in[0];
    const float* pe_fw    = (const float*)d_in[1];
    const float* pe_bw    = (const float*)d_in[2];
    const float* Wi_fw    = (const float*)d_in[3];
    const float* Wh_fw    = (const float*)d_in[4];
    const float* bi_fw    = (const float*)d_in[5];
    const float* bh_fw    = (const float*)d_in[6];
    const float* Wi_bw    = (const float*)d_in[7];
    const float* Wh_bw    = (const float*)d_in[8];
    const float* bi_bw    = (const float*)d_in[9];
    const float* bh_bw    = (const float*)d_in[10];
    float* out = (float*)d_out;

    char* ws = (char*)d_ws;
    __bf16* wpk  = (__bf16*)(ws + OFF_WPK);
    __bf16* xbf  = (__bf16*)(ws + OFF_XBF);
    __bf16* hbuf = (__bf16*)(ws + OFF_HBUF);
    float*  cbuf = (float*) (ws + OFF_CBUF);
    float*  bias = (float*) (ws + OFF_BIAS);

    { int n = (int)XBF_ELEMS;   pack_x_kernel<<<(n + 255)/256, 256, 0, stream>>>(features, xbf, n); }
    { int n = 2*2*1024*2048;    pack_w_kernel<<<(n + 255)/256, 256, 0, stream>>>(Wi_fw, Wh_fw, Wi_bw, Wh_bw, wpk); }
    { int n = 2*2*N_*H_;        init_state_kernel<<<(n + 255)/256, 256, 0, stream>>>(pe_fw, pe_bw, hbuf, cbuf); }
    { int n = 2*2*G4_;          make_bias_kernel<<<(n + 255)/256, 256, 0, stream>>>(bi_fw, bh_fw, bi_bw, bh_bw, bias); }

    dim3 grid(8, 16, 2);
    for (int s = 0; s < T_; ++s) {
        lstm_step_kernel<<<grid, 256, 0, stream>>>(wpk, xbf, hbuf, cbuf, bias, out, 0, s);
        lstm_step_kernel<<<grid, 256, 0, stream>>>(wpk, xbf, hbuf, cbuf, bias, out, 1, s);
    }
}
